// GAPL_84945863180510
// MI455X (gfx1250) — matrix operand order repaired and checked by reading, not compiled, not run on hardware
//
#include <hip/hip_runtime.h>
#include <hip/hip_bf16.h>

typedef __attribute__((ext_vector_type(16))) _Float16 v16h;
typedef __attribute__((ext_vector_type(8)))  float    v8f;

#define CCH 96
#define KLIN 224   // 198 padded to 7*32

// ---- monotonic float<->uint encoding for atomic max ----
__device__ __forceinline__ unsigned fenc(float x) {
  unsigned u = __float_as_uint(x);
  return (u & 0x80000000u) ? ~u : (u | 0x80000000u);
}
__device__ __forceinline__ float fdec(unsigned u) {
  u = (u & 0x80000000u) ? (u & 0x7FFFFFFFu) : ~u;
  return __uint_as_float(u);
}

// ---- global std of d=[x_j-x_i, pos_j-pos_i] (99 comps) and dp (3 comps) ----
__global__ void k_stats_d(const float* __restrict__ x, const float* __restrict__ pos,
                          const int* __restrict__ ei, int E,
                          double* __restrict__ dstats) {
  int total = E * 99;
  float sD = 0.f, qD = 0.f, sP = 0.f, qP = 0.f;
  for (int i = blockIdx.x * blockDim.x + threadIdx.x; i < total;
       i += gridDim.x * blockDim.x) {
    int e = i / 99, k = i - e * 99;
    int ii = ei[e], jj = ei[E + e];
    float d = (k < 96) ? (x[(size_t)jj * 96 + k] - x[(size_t)ii * 96 + k])
                       : (pos[jj * 3 + (k - 96)] - pos[ii * 3 + (k - 96)]);
    sD += d; qD += d * d;
    if (k >= 96) { sP += d; qP += d * d; }
  }
  for (int off = 16; off; off >>= 1) {
    sD += __shfl_xor(sD, off); qD += __shfl_xor(qD, off);
    sP += __shfl_xor(sP, off); qP += __shfl_xor(qP, off);
  }
  __shared__ float r[4];
  if (threadIdx.x < 4) r[threadIdx.x] = 0.f;
  __syncthreads();
  if ((threadIdx.x & 31) == 0) {
    atomicAdd(&r[0], sD); atomicAdd(&r[1], qD);
    atomicAdd(&r[2], sP); atomicAdd(&r[3], qP);
  }
  __syncthreads();
  if (threadIdx.x == 0) {
    atomicAdd(&dstats[0], (double)r[0]); atomicAdd(&dstats[1], (double)r[1]);
    atomicAdd(&dstats[2], (double)r[2]); atomicAdd(&dstats[3], (double)r[3]);
  }
}

__global__ void k_fin0(const double* __restrict__ dstats, float* __restrict__ scal, int E) {
  if (threadIdx.x == 0) {
    double nD = (double)E * 99.0;
    double vD = (dstats[1] - dstats[0] * dstats[0] / nD) / (nD - 1.0);
    scal[0] = (float)(1.0 / (sqrt(vD) + 1e-5));
    double nP = (double)E * 3.0;
    double vP = (dstats[3] - dstats[2] * dstats[2] / nP) / (nP - 1.0);
    scal[1] = (float)(1.0 / (sqrt(vP) + 1e-5));
  }
}

// ---- weight f32[K,96] -> f16 transposed [96][Kpad] (B-fragments contiguous) ----
__global__ void k_convw(const float* __restrict__ src, _Float16* __restrict__ dst,
                        int Kin, int Kpad) {
  int i = blockIdx.x * blockDim.x + threadIdx.x;
  if (i >= 96 * Kpad) return;
  int n = i / Kpad, k = i - n * Kpad;
  dst[i] = (k < Kin) ? (_Float16)src[(size_t)k * 96 + n] : (_Float16)0.f;
}

// ---- wave-level 16x96 tile GEMM via v_wmma_f32_16x16x32_f16 ----
template <int KT>
__device__ __forceinline__ void wmma_tile(const _Float16* __restrict__ As, int astr,
                                          const _Float16* __restrict__ Wg, int kpad,
                                          int rowBase, int ln, int hi, v8f* acc) {
#pragma unroll
  for (int nT = 0; nT < 6; ++nT) {
    v8f c = {};
    const _Float16* arow = As + (size_t)(rowBase + ln) * astr + 8 * hi;
    const _Float16* brow = Wg + (size_t)(nT * 16 + ln) * kpad + 8 * hi;
#pragma unroll
    for (int kt = 0; kt < KT; ++kt) {
      union { v16h v; uint4 q[2]; } A, B;
      A.q[0] = *(const uint4*)(arow + kt * 32);       // halves 0-7 : K = 32kt+8hi+0..7
      A.q[1] = *(const uint4*)(arow + kt * 32 + 16);  // halves 8-15: K = 32kt+16+8hi+0..7
      B.q[0] = *(const uint4*)(brow + kt * 32);       // K = 32kt+8hi+0..7
      B.q[1] = *(const uint4*)(brow + kt * 32 + 16);  // K = 32kt+16+8hi+0..7
      c = __builtin_amdgcn_wmma_f32_16x16x32_f16(false, A.v, false, B.v,
                                                 (short)0, c, false, false);
    }
    acc[nT] = c;
  }
}

// ---- epilogue: bias, store pre-BN, per-channel BN partial sums ----
__device__ __forceinline__ void epilogue(v8f* acc, const float* __restrict__ bias,
                                         float* __restrict__ out, int e0w, int E,
                                         int ln, int hi, float* sBlk, float* qBlk) {
#pragma unroll
  for (int nT = 0; nT < 6; ++nT) {
    int n = nT * 16 + ln;
    float b = bias[n];
    float s = 0.f, q = 0.f;
#pragma unroll
    for (int r = 0; r < 8; ++r) {
      int edge = e0w + r + 8 * hi;   // C/D layout: VGPR r -> M = r + 8*hi
      float v = acc[nT][r] + b;
      if (edge < E) { out[(size_t)edge * 96 + n] = v; s += v; q += v * v; }
    }
    s += __shfl_xor(s, 16);          // lanes L and L+16 share channel n
    q += __shfl_xor(q, 16);
    if (hi == 0) { atomicAdd(&sBlk[n], s); atomicAdd(&qBlk[n], q); }
  }
}

// ---- pass 2: gather + feature build + 198->96 GEMM + BN0 stats ----
__global__ void __launch_bounds__(256) k_lin_gemm(
    const float* __restrict__ x, const float* __restrict__ pos, const int* __restrict__ ei,
    const float* __restrict__ aw, const float* __restrict__ ab,
    const _Float16* __restrict__ Wg, const float* __restrict__ bias,
    const float* __restrict__ scal, float* __restrict__ out,
    double* __restrict__ chSum, double* __restrict__ chSsq, int E) {
  __shared__ _Float16 As[128 * KLIN];
  __shared__ float sBlk[96], qBlk[96];
  int tid = threadIdx.x;
  if (tid < 96) { sBlk[tid] = 0.f; qBlk[tid] = 0.f; }
  int e0 = blockIdx.x * 128;
  float invd = scal[0];
  for (int i = tid; i < 128 * KLIN; i += 256) {
    int le = i / KLIN, k = i - le * KLIN;
    int edge = e0 + le;
    float v = 0.f;
    if (edge < E && k < 198) {
      int ii = ei[edge], jj = ei[E + edge];
      if (k < 96) v = x[(size_t)ii * 96 + k];
      else if (k < 99) v = pos[ii * 3 + (k - 96)];
      else {
        int c = k - 99;
        float d = (c < 96) ? (x[(size_t)jj * 96 + c] - x[(size_t)ii * 96 + c])
                           : (pos[jj * 3 + (c - 96)] - pos[ii * 3 + (c - 96)]);
        v = aw[c] * (d * invd) + ab[c];
      }
    }
    As[(size_t)le * KLIN + k] = (_Float16)v;
  }
  __syncthreads();
  int w = tid >> 5, lane = tid & 31, ln = lane & 15, hi = lane >> 4;
  v8f acc[6];
  wmma_tile<7>(As, KLIN, Wg, KLIN, w * 16, ln, hi, acc);
  epilogue(acc, bias, out, e0 + w * 16, E, ln, hi, sBlk, qBlk);
  __syncthreads();
  if (tid < 96) {
    atomicAdd(&chSum[tid], (double)sBlk[tid]);
    atomicAdd(&chSsq[tid], (double)qBlk[tid]);
  }
}

// ---- mid passes: BN(+variant) -> f16 -> 96x96 GEMM -> next BN stats ----
// preMode: 0 = bn+relu then pe*(t+pe), store resid; 1 = bn+relu; 2 = bn+resid+relu, store resid
__global__ void __launch_bounds__(256) k_mid_gemm(
    int preMode, const float* __restrict__ in, float* __restrict__ resid,
    const float* __restrict__ mean, const float* __restrict__ rstd,
    const float* __restrict__ g, const float* __restrict__ be,
    const _Float16* __restrict__ Wg, const float* __restrict__ bias,
    float* __restrict__ out, double* __restrict__ chSum, double* __restrict__ chSsq,
    const float* __restrict__ pos, const int* __restrict__ ei,
    const float* __restrict__ scal, int E) {
  __shared__ _Float16 As[128 * 96];
  __shared__ float sBlk[96], qBlk[96];
  int tid = threadIdx.x;
  if (tid < 96) { sBlk[tid] = 0.f; qBlk[tid] = 0.f; }
  int e0 = blockIdx.x * 128;
  for (int i = tid; i < 128 * 96; i += 256) {
    int le = i / 96, c = i - le * 96;
    int edge = e0 + le;
    float v = 0.f;
    if (edge < E) {
      size_t gi = (size_t)edge * 96 + c;
      v = in[gi];
      v = (v - mean[c]) * rstd[c] * g[c] + be[c];
      if (preMode == 0) {
        v = fmaxf(v, 0.f);
        int ii = ei[edge], jj = ei[E + edge];
        int coord = c >> 5, rem = c & 31, f = rem >> 1;
        float pr = (pos[jj * 3 + coord] - pos[ii * 3 + coord]) * scal[1];
        // div = pr / 100^(f/16) = pr * exp(-f*ln(100)/16)
        float dv = pr * __expf(-0.2878231366242557f * (float)f);
        float pe = (rem & 1) ? __cosf(dv) : __sinf(dv);
        v = pe * (v + pe);
        resid[gi] = v;
      } else if (preMode == 1) {
        v = fmaxf(v, 0.f);
      } else {
        v = fmaxf(v + resid[gi], 0.f);
        resid[gi] = v;
      }
    }
    As[(size_t)le * 96 + c] = (_Float16)v;
  }
  __syncthreads();
  int w = tid >> 5, lane = tid & 31, ln = lane & 15, hi = lane >> 4;
  v8f acc[6];
  wmma_tile<3>(As, 96, Wg, 96, w * 16, ln, hi, acc);
  epilogue(acc, bias, out, e0 + w * 16, E, ln, hi, sBlk, qBlk);
  __syncthreads();
  if (tid < 96) {
    atomicAdd(&chSum[tid], (double)sBlk[tid]);
    atomicAdd(&chSsq[tid], (double)qBlk[tid]);
  }
}

__global__ void k_finbn(const double* __restrict__ chSum, const double* __restrict__ chSsq,
                        float* __restrict__ mean, float* __restrict__ rstd, int E) {
  int c = threadIdx.x;
  if (c < 96) {
    double m = chSum[c] / (double)E;
    double v = chSsq[c] / (double)E - m * m;  // biased var, matches jnp.var
    mean[c] = (float)m;
    rstd[c] = (float)(1.0 / sqrt(v + 1e-5));
  }
}

// ---- pass 7: final BN + residual + relu -> out_x; segment max (both temps) ----
__global__ void k_final_max(const float* __restrict__ in, float* __restrict__ resid,
                            const float* __restrict__ mean, const float* __restrict__ rstd,
                            const float* __restrict__ g, const float* __restrict__ be,
                            const int* __restrict__ ei, const float* __restrict__ tmax,
                            const float* __restrict__ tavg, unsigned* __restrict__ mEnc,
                            int E, int N) {
  int i = blockIdx.x * blockDim.x + threadIdx.x;
  if (i >= E * 96) return;
  int e = i / 96, c = i - e * 96;
  float v = in[i];
  v = (v - mean[c]) * rstd[c] * g[c] + be[c];
  v = fmaxf(v + resid[i], 0.f);
  resid[i] = v;  // out_x
  int node = ei[e];
  size_t o = (size_t)node * 96 + c;
  atomicMax(&mEnc[o], fenc(v * tmax[c]));
  atomicMax(&mEnc[(size_t)N * 96 + o], fenc(v * tavg[c]));
}

// ---- pass 8: e = exp(a-m); accumulate sum(e) and sum(out_x*e) ----
__global__ void k_agg_sum(const float* __restrict__ outx, const int* __restrict__ ei,
                          const float* __restrict__ tmax, const float* __restrict__ tavg,
                          const unsigned* __restrict__ mEnc, float* __restrict__ sAcc,
                          float* __restrict__ nAcc, int E, int N) {
  int i = blockIdx.x * blockDim.x + threadIdx.x;
  if (i >= E * 96) return;
  int e = i / 96, c = i - e * 96;
  float v = outx[i];
  int node = ei[e];
  size_t o = (size_t)node * 96 + c;
  size_t NC = (size_t)N * 96;
  float e1 = __expf(v * tmax[c] - fdec(mEnc[o]));
  atomicAdd(&sAcc[o], e1);      atomicAdd(&nAcc[o], v * e1);
  float e2 = __expf(v * tavg[c] - fdec(mEnc[NC + o]));
  atomicAdd(&sAcc[NC + o], e2); atomicAdd(&nAcc[NC + o], v * e2);
}

// ---- pass 9: out = num/(s+eps) summed over both temperatures ----
__global__ void k_combine(const float* __restrict__ sAcc, const float* __restrict__ nAcc,
                          float* __restrict__ out, int N) {
  int i = blockIdx.x * blockDim.x + threadIdx.x;
  if (i >= N * 96) return;
  size_t NC = (size_t)N * 96;
  out[i] = nAcc[i] / (sAcc[i] + 1e-16f) + nAcc[NC + i] / (sAcc[NC + i] + 1e-16f);
}

extern "C" void kernel_launch(void* const* d_in, const int* in_sizes, int n_in,
                              void* d_out, int out_size, void* d_ws, size_t ws_size,
                              hipStream_t stream) {
  const float* pos   = (const float*)d_in[0];
  const float* x     = (const float*)d_in[1];
  const int*   ei    = (const int*)d_in[2];
  const float* aw    = (const float*)d_in[3];
  const float* ab    = (const float*)d_in[4];
  const float* linW  = (const float*)d_in[5];
  const float* linb  = (const float*)d_in[6];
  const float* ling  = (const float*)d_in[7];
  const float* linbe = (const float*)d_in[8];
  const float* rW1   = (const float*)d_in[9];
  const float* rb1   = (const float*)d_in[10];
  const float* rg1   = (const float*)d_in[11];
  const float* rbe1  = (const float*)d_in[12];
  const float* rW2   = (const float*)d_in[13];
  const float* rb2   = (const float*)d_in[14];
  const float* rgo   = (const float*)d_in[15];
  const float* rbo   = (const float*)d_in[16];
  const float* tmax  = (const float*)d_in[17];
  const float* tavg  = (const float*)d_in[18];
  int N = in_sizes[0] / 3;
  int E = in_sizes[2] / 2;

  char* wsb = (char*)d_ws;
  double*   dstats = (double*)(wsb + 0);      // 4 doubles
  double*   chSum  = (double*)(wsb + 64);     // 5*96 doubles
  double*   chSsq  = (double*)(wsb + 4096);   // 5*96 doubles
  float*    scal   = (float*)(wsb + 8192);    // inv_std_d, inv_std_dp
  float*    bnMean = (float*)(wsb + 8448);    // 5*96
  float*    bnRstd = (float*)(wsb + 10368);   // 5*96
  _Float16* wLin   = (_Float16*)(wsb + 16384);   // 96*224
  _Float16* wRes   = (_Float16*)(wsb + 61440);   // 4 * 96*96
  float*    bufA   = (float*)(wsb + 139264);     // [E,96] GEMM chain (in-place)
  size_t EC = (size_t)E * 96;
  float*    bufR   = bufA + EC;                  // [E,96] residual / out_x
  unsigned* mEnc   = (unsigned*)(bufR + EC);     // 2*[N,96]
  size_t NC = (size_t)N * 96;
  float*    sAcc   = (float*)(mEnc + 2 * NC);    // 2*[N,96]
  float*    nAcc   = sAcc + 2 * NC;              // 2*[N,96]

  hipMemsetAsync(wsb, 0, 12288, stream);            // all reduction accumulators
  hipMemsetAsync(mEnc, 0, NC * 4 * 6, stream);      // mEnc(2)+sAcc(2)+nAcc(2), contiguous

  k_convw<<<(96 * KLIN + 255) / 256, 256, 0, stream>>>(linW, wLin, 198, KLIN);
  int cb2 = (96 * 96 + 255) / 256;
  k_convw<<<cb2, 256, 0, stream>>>(rW1,           wRes + 0 * 9216, 96, 96);
  k_convw<<<cb2, 256, 0, stream>>>(rW2,           wRes + 1 * 9216, 96, 96);
  k_convw<<<cb2, 256, 0, stream>>>(rW1 + 96 * 96, wRes + 2 * 9216, 96, 96);
  k_convw<<<cb2, 256, 0, stream>>>(rW2 + 96 * 96, wRes + 3 * 9216, 96, 96);

  k_stats_d<<<2048, 256, 0, stream>>>(x, pos, ei, E, dstats);
  k_fin0<<<1, 32, 0, stream>>>(dstats, scal, E);

  int gb = (E + 127) / 128;
  k_lin_gemm<<<gb, 256, 0, stream>>>(x, pos, ei, aw, ab, wLin, linb, scal, bufA,
                                     chSum + 0, chSsq + 0, E);
  k_finbn<<<1, 96, 0, stream>>>(chSum + 0, chSsq + 0, bnMean + 0, bnRstd + 0, E);

  k_mid_gemm<<<gb, 256, 0, stream>>>(0, bufA, bufR, bnMean + 0, bnRstd + 0, ling, linbe,
      wRes + 0 * 9216, rb1 + 0, bufA, chSum + 96, chSsq + 96, pos, ei, scal, E);
  k_finbn<<<1, 96, 0, stream>>>(chSum + 96, chSsq + 96, bnMean + 96, bnRstd + 96, E);

  k_mid_gemm<<<gb, 256, 0, stream>>>(1, bufA, bufR, bnMean + 96, bnRstd + 96, rg1 + 0, rbe1 + 0,
      wRes + 1 * 9216, rb2 + 0, bufA, chSum + 192, chSsq + 192, pos, ei, scal, E);
  k_finbn<<<1, 96, 0, stream>>>(chSum + 192, chSsq + 192, bnMean + 192, bnRstd + 192, E);

  k_mid_gemm<<<gb, 256, 0, stream>>>(2, bufA, bufR, bnMean + 192, bnRstd + 192, rgo + 0, rbo + 0,
      wRes + 2 * 9216, rb1 + 96, bufA, chSum + 288, chSsq + 288, pos, ei, scal, E);
  k_finbn<<<1, 96, 0, stream>>>(chSum + 288, chSsq + 288, bnMean + 288, bnRstd + 288, E);

  k_mid_gemm<<<gb, 256, 0, stream>>>(1, bufA, bufR, bnMean + 288, bnRstd + 288, rg1 + 96, rbe1 + 96,
      wRes + 3 * 9216, rb2 + 96, bufA, chSum + 384, chSsq + 384, pos, ei, scal, E);
  k_finbn<<<1, 96, 0, stream>>>(chSum + 384, chSsq + 384, bnMean + 384, bnRstd + 384, E);

  int eg = (int)((EC + 255) / 256);
  k_final_max<<<eg, 256, 0, stream>>>(bufA, bufR, bnMean + 384, bnRstd + 384, rgo + 96, rbo + 96,
                                      ei, tmax, tavg, mEnc, E, N);
  k_agg_sum<<<eg, 256, 0, stream>>>(bufR, ei, tmax, tavg, mEnc, sAcc, nAcc, E, N);
  k_combine<<<(int)((NC + 255) / 256), 256, 0, stream>>>(sAcc, nAcc, (float*)d_out, N);
}
